// SingleHeadAttention_9354438771094
// MI455X (gfx1250) — matrix operand order repaired and checked by reading, not compiled, not run on hardware
//
#include <hip/hip_runtime.h>
#include <hip/hip_bf16.h>

typedef __attribute__((ext_vector_type(16))) _Float16 v16h;
typedef __attribute__((ext_vector_type(8)))  _Float16 v8h;
typedef __attribute__((ext_vector_type(8)))  float    v8f;

#define EMBED_DIM 1024
#define ATTN_DIM  128
#define BATCH     4
#define SEQ       2048
#define SOFTMAX_SCALE 0.08838834764831845f   // 1/sqrt(128)

// A-fragment (16-bit, 16x32): lane-half holds K = half*16 + [0..15] here (contiguous order) (one 32B run)
// B-fragment (16-bit, 32x16): lane-half holds K = half*16 + [0..15] here (contiguous order) (one 32B run)
__device__ __forceinline__ v16h load_a_frag(const _Float16* p, int half) {
  return *(const v16h*)(p + half * 16);
}

// ---------------------------------------------------------------------------
// Phase 0: one-shot fp32 -> f16 conversion. 8 elements / thread, b128 stores.
// ---------------------------------------------------------------------------
__global__ __launch_bounds__(256) void cvt_f32_f16_kernel(
    const float* __restrict__ in, _Float16* __restrict__ out, int n8) {
  int i = blockIdx.x * 256 + threadIdx.x;
  if (i >= n8) return;
  const float4* p = (const float4*)in + (size_t)i * 2;
  float4 f0 = p[0];
  float4 f1 = p[1];
  v8h h;
  h[0] = (_Float16)f0.x; h[1] = (_Float16)f0.y;
  h[2] = (_Float16)f0.z; h[3] = (_Float16)f0.w;
  h[4] = (_Float16)f1.x; h[5] = (_Float16)f1.y;
  h[6] = (_Float16)f1.z; h[7] = (_Float16)f1.w;
  *((v8h*)out + i) = h;
}

// ---------------------------------------------------------------------------
// Phase 1: q/k/v = x @ W^T. 8 waves/block, 128 x-rows per block.
// W K-slab (32x128 f16, 8KB) staged in double-buffered LDS, shared by all
// waves; B-fragments read via ds_load_b128.
// ---------------------------------------------------------------------------
__global__ __launch_bounds__(256) void qkv_proj_kernel(
    const _Float16* __restrict__ x, const _Float16* __restrict__ W,
    _Float16* __restrict__ out, int transpose_out) {
  __shared__ alignas(32) _Float16 wl[2][128 * 32];   // [n][k] slabs, 8KB each

  const int tid  = threadIdx.x;
  const int wv   = tid >> 5;           // wave 0..7
  const int lane = tid & 31;
  const int half = lane >> 4;
  const int ln   = lane & 15;
  const int b    = blockIdx.x / (SEQ / 128);
  const int sb   = (blockIdx.x % (SEQ / 128)) * 128 + wv * 16;

  // staging role: thread copies one 32B chunk of the W slab
  const int sn   = tid >> 1;            // W row 0..127
  const int sseg = (tid & 1) * 16;      // k-offset 0 or 16

  v8f acc[8] = {};
  const _Float16* xrow = x + (size_t)(b * SEQ + sb + ln) * EMBED_DIM;

  // prologue: stage slab 0
  *(v16h*)(&wl[0][sn * 32 + sseg]) =
      *(const v16h*)(W + (size_t)sn * EMBED_DIM + sseg);
  __syncthreads();

  for (int it = 0; it < EMBED_DIM / 32; ++it) {
    const int kb  = it * 32;
    const int cur = it & 1;
    if (kb + 32 < EMBED_DIM)             // stage next slab into other buffer
      *(v16h*)(&wl[1 - cur][sn * 32 + sseg]) =
          *(const v16h*)(W + (size_t)sn * EMBED_DIM + kb + 32 + sseg);

    const v16h a = load_a_frag(xrow + kb, half);
#pragma unroll
    for (int nt = 0; nt < 8; ++nt) {
      const v16h bf = *(const v16h*)(&wl[cur][(nt * 16 + ln) * 32 + half * 16]);
      acc[nt] = __builtin_amdgcn_wmma_f32_16x16x32_f16(
          false, a, false, bf, (short)0, acc[nt], false, false);
    }
    __syncthreads();
  }

#pragma unroll
  for (int nt = 0; nt < 8; ++nt) {
#pragma unroll
    for (int r = 0; r < 8; ++r) {
      const int M = r + 8 * half;           // seq row within wave tile
      const int n = nt * 16 + ln;           // attn-dim column
      const _Float16 h = (_Float16)acc[nt][r];
      if (transpose_out)
        out[(size_t)(b * ATTN_DIM + n) * SEQ + sb + M] = h;
      else
        out[(size_t)(b * SEQ + sb + M) * ATTN_DIM + n] = h;
    }
  }
}

// ---------------------------------------------------------------------------
// Phase 2: flash attention. 8 waves/block covering 128 q rows; 32 keys/step.
// K tile (32x128) and V^T tile (128x32) staged in double-buffered LDS and
// shared by all 8 waves. Per wave: 8 score WMMAs + 8 PV WMMAs per step.
// ---------------------------------------------------------------------------
__global__ __launch_bounds__(256) void flash_attn_kernel(
    const _Float16* __restrict__ q, const _Float16* __restrict__ k,
    const _Float16* __restrict__ vT, float* __restrict__ out) {
  __shared__ alignas(32) _Float16 kl[2][32 * 128];   // [key][a], 8KB each
  __shared__ alignas(32) _Float16 vl[2][128 * 32];   // [a][key], 8KB each
  __shared__ alignas(32) _Float16 pl[8][16 * 32];    // per-wave P tile [M][K]

  const int tid  = threadIdx.x;
  const int wv   = tid >> 5;
  const int lane = tid & 31;
  const int half = lane >> 4;
  const int ln   = lane & 15;
  const int b    = blockIdx.x / (SEQ / 128);
  const int qb0  = (blockIdx.x % (SEQ / 128)) * 128;
  const int qb   = qb0 + wv * 16;      // this wave's 16 q rows

  const _Float16* kbase = k  + (size_t)b * SEQ * ATTN_DIM;
  const _Float16* vbase = vT + (size_t)b * ATTN_DIM * SEQ;

  // staging roles
  const int skey = tid >> 3;            // 0..31  (k tile row)
  const int ksg  = (tid & 7) * 16;      // 0..112 (k tile col chunk)
  const int sa   = tid >> 1;            // 0..127 (v tile row)
  const int vsg  = (tid & 1) * 16;      // 0 or 16

  // Q as 4 A-fragments (K = 0..127)
  v16h qf[4];
  const _Float16* qrow = q + (size_t)(b * SEQ + qb + ln) * ATTN_DIM;
#pragma unroll
  for (int s = 0; s < 4; ++s)
    qf[s] = load_a_frag(qrow + s * 32, half);

  float m[8], l[8];
  v8f o[8] = {};
#pragma unroll
  for (int r = 0; r < 8; ++r) { m[r] = -INFINITY; l[r] = 0.0f; }

  const int nIter = (qb0 + 128) / 32;   // keys 0 .. qb0+127 (block's last wave)

  // prologue: stage tile 0
  *(v16h*)(&kl[0][skey * 128 + ksg]) = *(const v16h*)(kbase + (size_t)skey * ATTN_DIM + ksg);
  *(v16h*)(&vl[0][sa * 32 + vsg])    = *(const v16h*)(vbase + (size_t)sa * SEQ + vsg);
  __syncthreads();

  for (int it = 0; it < nIter; ++it) {
    const int kb  = it * 32;
    const int cur = it & 1;
    if (it + 1 < nIter) {               // stage next tile into other buffer
      const int nkb = kb + 32;
      *(v16h*)(&kl[1 - cur][skey * 128 + ksg]) =
          *(const v16h*)(kbase + (size_t)(nkb + skey) * ATTN_DIM + ksg);
      *(v16h*)(&vl[1 - cur][sa * 32 + vsg]) =
          *(const v16h*)(vbase + (size_t)sa * SEQ + nkb + vsg);
    }

    if (kb <= qb + 15) {                // causal: this wave has work here
      // ---- scores: S = Q @ K^T (keys kb..+15 and kb+16..+31)
      v8f s0 = {}, s1 = {};
#pragma unroll
      for (int s = 0; s < 4; ++s) {
        const v16h b0 = *(const v16h*)(&kl[cur][(ln)      * 128 + s * 32 + half * 16]);
        const v16h b1 = *(const v16h*)(&kl[cur][(16 + ln) * 128 + s * 32 + half * 16]);
        s0 = __builtin_amdgcn_wmma_f32_16x16x32_f16(false, qf[s], false, b0, (short)0, s0, false, false);
        s1 = __builtin_amdgcn_wmma_f32_16x16x32_f16(false, qf[s], false, b1, (short)0, s1, false, false);
      }

      // ---- online softmax (row M = r + 8*half spans 16 lanes of one half)
      float alpha[8];
#pragma unroll
      for (int r = 0; r < 8; ++r) {
        const int M = r + 8 * half;
        float v0 = s0[r] * SOFTMAX_SCALE;
        float v1 = s1[r] * SOFTMAX_SCALE;
        if (kb + ln      > qb + M) v0 = -INFINITY;
        if (kb + 16 + ln > qb + M) v1 = -INFINITY;
        float mx = fmaxf(v0, v1);
#pragma unroll
        for (int d = 1; d < 16; d <<= 1)
          mx = fmaxf(mx, __shfl_xor(mx, d, 32));
        const float mnew = fmaxf(m[r], mx);
        const float a_   = __expf(m[r] - mnew);
        const float p0   = __expf(v0 - mnew);
        const float p1   = __expf(v1 - mnew);
        float rs = p0 + p1;
#pragma unroll
        for (int d = 1; d < 16; d <<= 1)
          rs += __shfl_xor(rs, d, 32);
        m[r] = mnew;
        l[r] = l[r] * a_ + rs;
        alpha[r] = a_;
        pl[wv][M * 32 + ln]      = (_Float16)p0;   // C-frag -> A-frag bounce
        pl[wv][M * 32 + 16 + ln] = (_Float16)p1;
      }

      const v16h pf = load_a_frag(&pl[wv][ln * 32], half);

      // ---- O = O*alpha + P @ V
#pragma unroll
      for (int nt = 0; nt < 8; ++nt) {
#pragma unroll
        for (int r = 0; r < 8; ++r) o[nt][r] *= alpha[r];
        const v16h vf = *(const v16h*)(&vl[cur][(nt * 16 + ln) * 32 + half * 16]);
        o[nt] = __builtin_amdgcn_wmma_f32_16x16x32_f16(false, pf, false, vf, (short)0, o[nt], false, false);
      }
    }
    __syncthreads();                    // uniform across all waves
  }

  // ---- epilogue: normalize and store fp32
#pragma unroll
  for (int nt = 0; nt < 8; ++nt) {
#pragma unroll
    for (int r = 0; r < 8; ++r) {
      const int M = r + 8 * half;
      out[(size_t)(b * SEQ + qb + M) * ATTN_DIM + nt * 16 + ln] = o[nt][r] / l[r];
    }
  }
}

// ---------------------------------------------------------------------------
extern "C" void kernel_launch(void* const* d_in, const int* in_sizes, int n_in,
                              void* d_out, int out_size, void* d_ws, size_t ws_size,
                              hipStream_t stream) {
  // setup_inputs() order: embedded, Wk, Wq, Wv
  const float* x  = (const float*)d_in[0];
  const float* Wk = (const float*)d_in[1];
  const float* Wq = (const float*)d_in[2];
  const float* Wv = (const float*)d_in[3];

  const size_t x_elems = (size_t)BATCH * SEQ * EMBED_DIM;      // 8.39M
  const size_t w_elems = (size_t)ATTN_DIM * EMBED_DIM;         // 131K
  const size_t p_elems = (size_t)BATCH * SEQ * ATTN_DIM;       // 1.05M

  _Float16* x16  = (_Float16*)d_ws;
  _Float16* wq16 = x16 + x_elems;
  _Float16* wk16 = wq16 + w_elems;
  _Float16* wv16 = wk16 + w_elems;
  _Float16* qbuf = wv16 + w_elems;
  _Float16* kbuf = qbuf + p_elems;
  _Float16* vbuf = kbuf + p_elems;                             // transposed [B][A][S]

  // Phase 0: fp32 -> f16 (x once, each W once)
  {
    int n8 = (int)(x_elems / 8);
    cvt_f32_f16_kernel<<<(n8 + 255) / 256, 256, 0, stream>>>(x, x16, n8);
    n8 = (int)(w_elems / 8);
    cvt_f32_f16_kernel<<<(n8 + 255) / 256, 256, 0, stream>>>(Wq, wq16, n8);
    cvt_f32_f16_kernel<<<(n8 + 255) / 256, 256, 0, stream>>>(Wk, wk16, n8);
    cvt_f32_f16_kernel<<<(n8 + 255) / 256, 256, 0, stream>>>(Wv, wv16, n8);
  }

  dim3 grid(BATCH * (SEQ / 128));   // 64 blocks x 8 waves
  qkv_proj_kernel<<<grid, 256, 0, stream>>>(x16, wq16, qbuf, 0);
  qkv_proj_kernel<<<grid, 256, 0, stream>>>(x16, wk16, kbuf, 0);
  qkv_proj_kernel<<<grid, 256, 0, stream>>>(x16, wv16, vbuf, 1);
  flash_attn_kernel<<<grid, 256, 0, stream>>>(qbuf, kbuf, vbuf, (float*)d_out);
}
